// CascadeGradNetOURS_86122684220072
// MI455X (gfx1250) — compile-verified, matrix operand order checked by reading, not run on hardware
//
#include <hip/hip_runtime.h>
#include <cstdint>
#include <cstddef>

typedef __attribute__((ext_vector_type(16))) __bf16 bf16x16;
typedef __attribute__((ext_vector_type(8)))  __bf16 bf16x8;
typedef __attribute__((ext_vector_type(8)))  float  f32x8;
typedef __attribute__((ext_vector_type(4)))  int    v4i;

#if __has_builtin(__builtin_amdgcn_global_load_async_to_lds_b128)
#define USE_ASYNC_LDS 1
#else
#define USE_ASYNC_LDS 0
#endif

namespace {
constexpr int kB  = 8192;   // batch rows
constexpr int kIN = 1024;   // in_dim
constexpr int kE  = 4096;   // embed_dim
constexpr int kL  = 8;      // layers

constexpr int KSTEP = 32;    // K per staged slab == one WMMA k-step
constexpr int LDSPITCH = 40; // padded LDS row stride (elements): bank stride 20
                             // -> 16 fragment rows hit distinct 4-bank groups

// workspace layout (bytes)
constexpr size_t OFF_XH = 0;                                  // bf16 x   (B  x IN) 16 MiB
constexpr size_t OFF_WB = OFF_XH + (size_t)kB  * kIN * 2;     // bf16 W   (E  x IN)  8 MiB
constexpr size_t OFF_WT = OFF_WB + (size_t)kE  * kIN * 2;     // bf16 W^T (IN x E )  8 MiB
constexpr size_t OFF_Z  = OFF_WT + (size_t)kIN * kE  * 2;     // bf16 z   (B  x E ) 64 MiB
} // namespace

// fp32 -> bf16 round-to-nearest-even
__device__ __forceinline__ __bf16 f2bf(float f) {
  unsigned int u = __builtin_bit_cast(unsigned int, f);
  unsigned int lsb = (u >> 16) & 1u;
  u += 0x7fffu + lsb;
  unsigned short h = (unsigned short)(u >> 16);
  return __builtin_bit_cast(__bf16, h);
}

// ---- async copy of one 16-byte chunk global -> LDS --------------------------
__device__ __forceinline__ void async_copy16(const __bf16* g, __bf16* l) {
#if USE_ASYNC_LDS
  __builtin_amdgcn_global_load_async_to_lds_b128(
      (__attribute__((address_space(1))) v4i*)g,
      (__attribute__((address_space(3))) v4i*)l, 0, 0);
#else
  *(bf16x8*)l = *(const bf16x8*)g;   // sync fallback: global load + ds store
#endif
}

__device__ __forceinline__ void wait_async_done() {
#if USE_ASYNC_LDS
#if __has_builtin(__builtin_amdgcn_s_wait_asynccnt)
  __builtin_amdgcn_s_wait_asynccnt(0);
#else
  asm volatile("s_wait_asynccnt 0x0" ::: "memory");
#endif
#endif
}

// Stage a ROWS x 32 bf16 slab (K-major, row stride ld) into LDS (pitch LDSPITCH).
// 256 threads; each thread moves ROWS*4/256 16-byte chunks.
template <int ROWS>
__device__ __forceinline__ void stage_slab(const __bf16* __restrict__ g, int ld,
                                           __bf16* l, int tid) {
#pragma unroll
  for (int i = 0; i < ROWS * 4 / 256; ++i) {
    int c   = tid + i * 256;
    int row = c >> 2;
    int kc  = (c & 3) * 8;
    async_copy16(g + (size_t)row * ld + kc, l + row * LDSPITCH + kc);
  }
}

// Read one 16x32 bf16 WMMA operand fragment from an LDS slab.
// lanes 0-15: row M=lane, K={0..7,16..23}; lanes 16-31: row M=lane-16, K={8..15,24..31}
__device__ __forceinline__ bf16x16 lds_frag(const __bf16* base, int lane) {
  int r  = lane & 15;
  int kb = (lane & 16) ? 8 : 0;
  const __bf16* q = base + r * LDSPITCH + kb;
  union { bf16x16 v; bf16x8 h[2]; } u;
  u.h[0] = *(const bf16x8*)(q);
  u.h[1] = *(const bf16x8*)(q + 16);
  return u.v;
}

// ---------------- prep kernels ----------------

__global__ __launch_bounds__(256) void cvt_f32_to_bf16(
    const float* __restrict__ in, __bf16* __restrict__ out, int n) {
  int i = blockIdx.x * 256 + threadIdx.x;
  if (i < n) out[i] = f2bf(in[i]);
}

// Wt[i*E + e] = bf16( W[e*IN + i] )
__global__ __launch_bounds__(256) void transpose_w(
    const float* __restrict__ W, __bf16* __restrict__ Wt) {
  int idx = blockIdx.x * 256 + threadIdx.x;   // over IN*E
  if (idx < kIN * kE) {
    int e = idx % kE;
    int i = idx / kE;
    Wt[idx] = f2bf(W[(size_t)e * kIN + i]);
  }
}

// ---------------- GEMM1 + fused cascade ----------------
// h[b,e] = sum_i xh[b,i]*Wb[e,i]; z = cascade(h) -> zb (bf16)
// WG: 256 thr = 8 waves (2 M x 4 N); WG tile 128x256; wave tile 64x64.
__global__ __launch_bounds__(256) void gemm1_cascade(
    const __bf16* __restrict__ xh, const __bf16* __restrict__ Wb,
    const float* __restrict__ biases, const float* __restrict__ alpha,
    const float* __restrict__ beta, __bf16* __restrict__ zb) {
  __shared__ alignas(16) __bf16 sA[2][128 * LDSPITCH];
  __shared__ alignas(16) __bf16 sB[2][256 * LDSPITCH];

  const int tid  = threadIdx.x;
  const int lane = tid & 31;
  const int wave = tid >> 5;
  const int wm = wave & 1;        // 0..1
  const int wn = wave >> 1;       // 0..3
  const int m0 = blockIdx.y * 128;
  const int n0 = blockIdx.x * 256;

  const __bf16* Ag = xh + (size_t)m0 * kIN;
  const __bf16* Bg = Wb + (size_t)n0 * kIN;

  f32x8 acc[4][4] = {};

  stage_slab<128>(Ag, kIN, sA[0], tid);
  stage_slab<256>(Bg, kIN, sB[0], tid);

  constexpr int NK = kIN / KSTEP;          // 32
  for (int ks = 0; ks < NK; ++ks) {
    const int cur = ks & 1;
    wait_async_done();        // my writes to sX[cur] have landed
    __syncthreads();          // everyone's writes landed; prev reads finished
    if (ks + 1 < NK) {
      stage_slab<128>(Ag + (ks + 1) * KSTEP, kIN, sA[cur ^ 1], tid);
      stage_slab<256>(Bg + (ks + 1) * KSTEP, kIN, sB[cur ^ 1], tid);
    }
    bf16x16 af[4], bfr[4];
#pragma unroll
    for (int tm = 0; tm < 4; ++tm)
      af[tm] = lds_frag(&sA[cur][(64 * wm + 16 * tm) * LDSPITCH], lane);
#pragma unroll
    for (int tn = 0; tn < 4; ++tn)
      bfr[tn] = lds_frag(&sB[cur][(64 * wn + 16 * tn) * LDSPITCH], lane);
#pragma unroll
    for (int tm = 0; tm < 4; ++tm)
#pragma unroll
      for (int tn = 0; tn < 4; ++tn)
        acc[tm][tn] = __builtin_amdgcn_wmma_f32_16x16x32_bf16(
            false, af[tm], false, bfr[tn], (short)0, acc[tm][tn], false, false);
  }

  // Cascade epilogue; column e fixed per lane within a tile column.
  const int mofs = (lane & 16) ? 8 : 0;
#pragma unroll
  for (int tn = 0; tn < 4; ++tn) {
    const int e = n0 + 64 * wn + 16 * tn + (lane & 15);
    float al[kL], be[kL], bi[kL];
#pragma unroll
    for (int l = 0; l < kL; ++l) {
      al[l] = alpha[l * kE + e];
      be[l] = beta[l * kE + e];
      bi[l] = biases[l * kE + e];
    }
#pragma unroll
    for (int tm = 0; tm < 4; ++tm) {
#pragma unroll
      for (int r = 0; r < 8; ++r) {
        const float h = acc[tm][tn][r];
        float z = be[0] * (h + bi[0]);
#pragma unroll
        for (int i = 0; i < kL - 1; ++i)
          z = be[i + 1] * (h + bi[i + 1]) + al[i] * fmaxf(z, 0.0f);
        z = al[kL - 1] * fmaxf(z, 0.0f);
        const int m = m0 + 64 * wm + 16 * tm + r + mofs;
        zb[(size_t)m * kE + e] = f2bf(z);
      }
    }
  }
}

// ---------------- GEMM2 ----------------
// out[b,i] = sum_e zb[b,e]*Wt[i,e] + bias_last[i]
__global__ __launch_bounds__(256) void gemm2(
    const __bf16* __restrict__ zb, const __bf16* __restrict__ Wt,
    const float* __restrict__ bias_last, float* __restrict__ out) {
  __shared__ alignas(16) __bf16 sA[2][128 * LDSPITCH];
  __shared__ alignas(16) __bf16 sB[2][256 * LDSPITCH];

  const int tid  = threadIdx.x;
  const int lane = tid & 31;
  const int wave = tid >> 5;
  const int wm = wave & 1;
  const int wn = wave >> 1;
  const int m0 = blockIdx.y * 128;
  const int n0 = blockIdx.x * 256;

  const __bf16* Ag = zb + (size_t)m0 * kE;
  const __bf16* Bg = Wt + (size_t)n0 * kE;

  f32x8 acc[4][4] = {};

  stage_slab<128>(Ag, kE, sA[0], tid);
  stage_slab<256>(Bg, kE, sB[0], tid);

  constexpr int NK = kE / KSTEP;           // 128
  for (int ks = 0; ks < NK; ++ks) {
    const int cur = ks & 1;
    wait_async_done();
    __syncthreads();
    if (ks + 1 < NK) {
      stage_slab<128>(Ag + (ks + 1) * KSTEP, kE, sA[cur ^ 1], tid);
      stage_slab<256>(Bg + (ks + 1) * KSTEP, kE, sB[cur ^ 1], tid);
    }
    bf16x16 af[4], bfr[4];
#pragma unroll
    for (int tm = 0; tm < 4; ++tm)
      af[tm] = lds_frag(&sA[cur][(64 * wm + 16 * tm) * LDSPITCH], lane);
#pragma unroll
    for (int tn = 0; tn < 4; ++tn)
      bfr[tn] = lds_frag(&sB[cur][(64 * wn + 16 * tn) * LDSPITCH], lane);
#pragma unroll
    for (int tm = 0; tm < 4; ++tm)
#pragma unroll
      for (int tn = 0; tn < 4; ++tn)
        acc[tm][tn] = __builtin_amdgcn_wmma_f32_16x16x32_bf16(
            false, af[tm], false, bfr[tn], (short)0, acc[tm][tn], false, false);
  }

  const int mofs = (lane & 16) ? 8 : 0;
#pragma unroll
  for (int tn = 0; tn < 4; ++tn) {
    const int n = n0 + 64 * wn + 16 * tn + (lane & 15);
    const float bl = bias_last[n];
#pragma unroll
    for (int tm = 0; tm < 4; ++tm) {
#pragma unroll
      for (int r = 0; r < 8; ++r) {
        const int m = m0 + 64 * wm + 16 * tm + r + mofs;
        out[(size_t)m * kIN + n] = acc[tm][tn][r] + bl;
      }
    }
  }
}

// ---------------- launch ----------------

extern "C" void kernel_launch(void* const* d_in, const int* in_sizes, int n_in,
                              void* d_out, int out_size, void* d_ws,
                              size_t ws_size, hipStream_t stream) {
  const float* x         = (const float*)d_in[0];  // (B, IN)
  const float* W         = (const float*)d_in[1];  // (E, IN)
  const float* biases    = (const float*)d_in[2];  // (L+1, E)
  const float* bias_last = (const float*)d_in[3];  // (IN,)
  const float* alpha     = (const float*)d_in[4];  // (L, E)
  const float* beta      = (const float*)d_in[5];  // (L, E)
  float* out = (float*)d_out;

  char* ws = (char*)d_ws;
  __bf16* xh = (__bf16*)(ws + OFF_XH);
  __bf16* Wb = (__bf16*)(ws + OFF_WB);
  __bf16* Wt = (__bf16*)(ws + OFF_WT);
  __bf16* zb = (__bf16*)(ws + OFF_Z);

  {
    int n = kB * kIN;
    cvt_f32_to_bf16<<<(n + 255) / 256, 256, 0, stream>>>(x, xh, n);
  }
  {
    int n = kE * kIN;
    cvt_f32_to_bf16<<<(n + 255) / 256, 256, 0, stream>>>(W, Wb, n);
    transpose_w<<<(n + 255) / 256, 256, 0, stream>>>(W, Wt);
  }

  gemm1_cascade<<<dim3(kE / 256, kB / 128), 256, 0, stream>>>(
      xh, Wb, biases, alpha, beta, zb);

  gemm2<<<dim3(kIN / 256, kB / 128), 256, 0, stream>>>(
      zb, Wt, bias_last, out);

  (void)in_sizes; (void)n_in; (void)out_size; (void)ws_size;
}
